// MeshAutoencoder_64381559767559
// MI455X (gfx1250) — compile-verified, matrix operand order checked by reading, not run on hardware
//
#include <hip/hip_runtime.h>
#include <hip/hip_bf16.h>

typedef __attribute__((ext_vector_type(16))) _Float16 v16h;
typedef __attribute__((ext_vector_type(8)))  float    v8f;

#define PI_F 3.14159265358979323846f

// ---------------------------------------------------------------------------
// helpers
// ---------------------------------------------------------------------------
__device__ __forceinline__ int discretize128(float t, float lo, float hi) {
    float u = (t - lo) / (hi - lo) * 128.0f - 0.5f;
    int r = (int)rintf(u);
    return r < 0 ? 0 : (r > 127 ? 127 : r);
}

__device__ __forceinline__ float silu_f(float v) {
    return v / (1.0f + __expf(-v));
}

// ---------------------------------------------------------------------------
// 1) per-face geometry -> 16 discrete indices
//    layout: [0..8] coor(v-major,xyz) [9..11] angle [12..14] normal [15] area
// ---------------------------------------------------------------------------
__global__ void face_feat_idx(const float* __restrict__ vertices,
                              const int* __restrict__ faces,
                              int* __restrict__ idx,
                              int nb, int nv, int nf) {
    int f = blockIdx.x * blockDim.x + threadIdx.x;
    int total = nb * nf;
    if (f >= total) return;
    int b = f / nf;

    float fc[3][3];
    for (int v = 0; v < 3; ++v) {
        int vi = faces[(size_t)f * 3 + v];
        const float* p = vertices + ((size_t)b * nv + vi) * 3;
        fc[v][0] = p[0]; fc[v][1] = p[1]; fc[v][2] = p[2];
    }
    int* o = idx + (size_t)f * 16;
    // coordinates
    for (int v = 0; v < 3; ++v)
        for (int d = 0; d < 3; ++d)
            o[v * 3 + d] = discretize128(fc[v][d], -1.0f, 1.0f);
    // angles: shifted = roll(+1) -> pairs (0,2),(1,0),(2,1)
    for (int v = 0; v < 3; ++v) {
        const float* a = fc[v];
        const float* s = fc[(v + 2) % 3];
        float na = fmaxf(sqrtf(a[0]*a[0] + a[1]*a[1] + a[2]*a[2]), 1e-12f);
        float ns = fmaxf(sqrtf(s[0]*s[0] + s[1]*s[1] + s[2]*s[2]), 1e-12f);
        float c = (a[0]*s[0] + a[1]*s[1] + a[2]*s[2]) / (na * ns);
        c = fminf(fmaxf(c, -1.0f + 1e-5f), 1.0f - 1e-5f);
        o[9 + v] = discretize128(acosf(c), 0.0f, PI_F);
    }
    // edges, cross, normal, area
    float e0[3], e1[3], cr[3];
    for (int d = 0; d < 3; ++d) { e0[d] = fc[0][d] - fc[2][d]; e1[d] = fc[1][d] - fc[0][d]; }
    cr[0] = e0[1]*e1[2] - e0[2]*e1[1];
    cr[1] = e0[2]*e1[0] - e0[0]*e1[2];
    cr[2] = e0[0]*e1[1] - e0[1]*e1[0];
    float cn = sqrtf(cr[0]*cr[0] + cr[1]*cr[1] + cr[2]*cr[2]);
    float inv = 1.0f / fmaxf(cn, 1e-12f);
    for (int d = 0; d < 3; ++d)
        o[12 + d] = discretize128(cr[d] * inv, -1.0f, 1.0f);
    o[15] = discretize128(cn * 0.5f, 0.0f, 4.0f);
}

// ---------------------------------------------------------------------------
// 2) materialize 832-d embedding rows as f16 (GEMM A operand)
// ---------------------------------------------------------------------------
__global__ void build_feat(const int* __restrict__ idx,
                           const float* __restrict__ coor_tab,
                           const float* __restrict__ angle_tab,
                           const float* __restrict__ normal_tab,
                           const float* __restrict__ area_tab,
                           _Float16* __restrict__ feat, int M) {
    size_t i = (size_t)blockIdx.x * blockDim.x + threadIdx.x;
    size_t total = (size_t)M * 832;
    if (i >= total) return;
    int m = (int)(i / 832);
    int k = (int)(i % 832);
    const int* o = idx + (size_t)m * 16;
    float v;
    if (k < 576)       v = coor_tab[o[k >> 6] * 64 + (k & 63)];
    else if (k < 624)  { int kk = k - 576; v = angle_tab[o[9 + (kk >> 4)] * 16 + (kk & 15)]; }
    else if (k < 816)  { int kk = k - 624; v = normal_tab[o[12 + (kk >> 6)] * 64 + (kk & 63)]; }
    else               v = area_tab[o[15] * 16 + (k - 816)];
    feat[i] = (_Float16)v;
}

// ---------------------------------------------------------------------------
// 3) pack conv/proj weights into exact WMMA B-fragment order (f16)
//    fragment index = ((t*kcn+kc)*cot + coTile) ; 32 lanes x 16 halves each
//    src_is_kn=1: src layout [K][N] (proj_w) ; else [co][ci][t] (conv weights)
// ---------------------------------------------------------------------------
__global__ void prep_w_packed(const float* __restrict__ w, _Float16* __restrict__ o,
                              int Cout, int Cin, int KT, int src_is_kn) {
    size_t total = (size_t)KT * Cin * Cout;
    size_t i = (size_t)blockIdx.x * blockDim.x + threadIdx.x;
    if (i >= total) return;
    int j    = (int)(i & 15);
    int lane = (int)((i >> 4) & 31);
    size_t rest = i >> 9;
    int cot = Cout >> 4, kcn = Cin >> 5;
    int coTile = (int)(rest % cot); rest /= cot;
    int kc = (int)(rest % kcn);
    int t  = (int)(rest / kcn);
    int n = lane & 15;
    int koffL = (lane < 16) ? 0 : 8;
    int k_local = (j < 8) ? (koffL + j) : (16 + koffL + (j - 8));
    int k  = kc * 32 + k_local;
    int co = coTile * 16 + n;
    float v = src_is_kn ? w[(size_t)k * Cout + co]
                        : w[((size_t)co * Cin + k) * KT + t];
    o[i] = (_Float16)v;
}

// ---------------------------------------------------------------------------
// 4) conv1d / GEMM via WMMA: y[b][n][co] = sum_{t,ci} x[b][n+t-pad][ci]*W
//    x: [b][pos][Cin] f16 ; wpk: WMMA-packed ; y: [b][pos][Cout] f32 (+bias)
//    One wave computes a 16(pos) x 64(co) tile with 4 independent
//    accumulators; K unrolled by 2 (all Cin/32 counts are even).
//    4 waves / block cover 64 positions.
// ---------------------------------------------------------------------------
__global__ void conv_gemm_wmma(const _Float16* __restrict__ x,
                               const _Float16* __restrict__ wpk,
                               const float* __restrict__ bias,
                               float* __restrict__ y,
                               int Npos, int Cin, int Cout, int KT, int pad) {
    const int lane = threadIdx.x & 31;
    const int wid  = threadIdx.x >> 5;
    const int n0   = (blockIdx.x * 4 + wid) * 16;
    const int coT0 = blockIdx.y * 4;         // 4 consecutive 16-wide N tiles
    const int b    = blockIdx.z;
    if (n0 >= Npos) return;                  // wave-uniform

    const int mrow = lane & 15;
    const int koff = (lane < 16) ? 0 : 8;
    const int kcn  = Cin >> 5;
    const int cot  = Cout >> 4;

    const _Float16* xb = x + (size_t)b * Npos * Cin;
    const v16h* wv = (const v16h*)wpk;

    v8f acc0 = {}, acc1 = {}, acc2 = {}, acc3 = {};

    for (int t = 0; t < KT; ++t) {
        const int row = n0 + mrow + t - pad;
        const bool valid = (row >= 0) && (row < Npos);
        const _Float16* arow = xb + (size_t)(valid ? row : 0) * Cin;

        for (int kc = 0; kc < kcn; kc += 2) {
            #pragma unroll
            for (int u = 0; u < 2; ++u) {
                v16h av;
                if (valid) {
                    const _Float16* p = arow + (kc + u) * 32 + koff;
                    #pragma unroll
                    for (int j = 0; j < 8; ++j) { av[j] = p[j]; av[8 + j] = p[16 + j]; }
                } else {
                    #pragma unroll
                    for (int j = 0; j < 16; ++j) av[j] = (_Float16)0.0f;
                }
                const size_t base =
                    ((size_t)(t * kcn + kc + u) * cot + coT0) * 32 + lane;
                v16h bv0 = wv[base];
                v16h bv1 = wv[base + 32];
                v16h bv2 = wv[base + 64];
                v16h bv3 = wv[base + 96];
                acc0 = __builtin_amdgcn_wmma_f32_16x16x32_f16(
                           false, av, false, bv0, (short)0, acc0, false, false);
                acc1 = __builtin_amdgcn_wmma_f32_16x16x32_f16(
                           false, av, false, bv1, (short)0, acc1, false, false);
                acc2 = __builtin_amdgcn_wmma_f32_16x16x32_f16(
                           false, av, false, bv2, (short)0, acc2, false, false);
                acc3 = __builtin_amdgcn_wmma_f32_16x16x32_f16(
                           false, av, false, bv3, (short)0, acc3, false, false);
            }
            // pull next K-step's weight fragments toward the WGP while the
            // matrix pipe is busy (lowers to global_prefetch_b8)
            if (kc + 2 < kcn) {
                const size_t nb_ = ((size_t)(t * kcn + kc + 2) * cot + coT0) * 32 + lane;
                __builtin_prefetch((const void*)&wv[nb_], 0, 1);
            }
        }
    }

    float* yb = y + (size_t)b * Npos * Cout;
    const int mbase = n0 + ((lane < 16) ? 0 : 8);
    #pragma unroll
    for (int q = 0; q < 4; ++q) {
        const int col = (coT0 + q) * 16 + mrow;
        const float bb = bias ? bias[col] : 0.0f;
        const v8f acc = (q == 0) ? acc0 : (q == 1) ? acc1 : (q == 2) ? acc2 : acc3;
        #pragma unroll
        for (int r = 0; r < 8; ++r)
            yb[(size_t)(mbase + r) * Cout + col] = acc[r] + bb;
    }
}

// ---------------------------------------------------------------------------
// 5) scatter-mean onto vertices (sum + count), then divide+gather per face
// ---------------------------------------------------------------------------
__global__ void zero_f32(float* p, size_t n) {
    size_t i = (size_t)blockIdx.x * blockDim.x + threadIdx.x;
    if (i < n) p[i] = 0.0f;
}

__global__ void scatter_add(const float* __restrict__ fe, const int* __restrict__ faces,
                            float* __restrict__ vsum, float* __restrict__ cnt,
                            int nb, int nv, int nf, int d) {
    size_t total = (size_t)nb * nf * 3 * d;
    size_t i = (size_t)blockIdx.x * blockDim.x + threadIdx.x;
    if (i >= total) return;
    int c = (int)(i % d);
    size_t r = i / d;
    int v = (int)(r % 3);
    size_t f = r / 3;          // global face index (b*nf+fi)
    int b = (int)(f / nf);
    int vi = faces[f * 3 + v] + b * nv;
    atomicAdd(&vsum[(size_t)vi * d + c], fe[f * d + c]);
    if (c == 0) atomicAdd(&cnt[vi], 1.0f);
}

__global__ void gather_vfeat(const float* __restrict__ vsum, const float* __restrict__ cnt,
                             const int* __restrict__ faces, _Float16* __restrict__ out,
                             int nb, int nv, int nf, int d) {
    size_t total = (size_t)nb * nf * 3 * d;
    size_t i = (size_t)blockIdx.x * blockDim.x + threadIdx.x;
    if (i >= total) return;
    int c = (int)(i % d);
    size_t r = i / d;
    int v = (int)(r % 3);
    size_t f = r / 3;
    int b = (int)(f / nf);
    int vi = faces[f * 3 + v] + b * nv;
    float val = vsum[(size_t)vi * d + c] / fmaxf(cnt[vi], 1e-5f);
    out[(f * 3 + v) * d + c] = (_Float16)val;   // [face][3*d] f16
}

// ---------------------------------------------------------------------------
// 6) (optional pixel-norm) + SiLU ; writes f32 and f16 activation copies
//    one block (256 thr) per position, block reduction over channels
// ---------------------------------------------------------------------------
__global__ void norm_silu(const float* __restrict__ y,
                          float* __restrict__ of32, _Float16* __restrict__ of16,
                          int C, int do_pn) {
    const int pos = blockIdx.x;
    const float* yp = y + (size_t)pos * C;
    __shared__ float red[256];
    float ss = 0.0f;
    for (int c = threadIdx.x; c < C; c += blockDim.x) { float v = yp[c]; ss += v * v; }
    red[threadIdx.x] = ss;
    __syncthreads();
    for (int s = 128; s > 0; s >>= 1) {
        if (threadIdx.x < s) red[threadIdx.x] += red[threadIdx.x + s];
        __syncthreads();
    }
    float scale = 1.0f;
    if (do_pn) scale = sqrtf((float)C) / fmaxf(sqrtf(red[0]), 1e-4f);
    for (int c = threadIdx.x; c < C; c += blockDim.x) {
        float s = silu_f(yp[c] * scale);
        of32[(size_t)pos * C + c] = s;
        of16[(size_t)pos * C + c] = (_Float16)s;
    }
}

// ---------------------------------------------------------------------------
// 7) SE path: two-stage channel mean, tiny 2-layer gate, scale+residual
// ---------------------------------------------------------------------------
#define MEAN_CHUNKS 32

__global__ void mean_partial(const float* __restrict__ h, float* __restrict__ part,
                             int Npos, int C) {
    int c  = blockIdx.x * 64 + threadIdx.x;
    int b  = blockIdx.y;
    int nbatch = gridDim.y;
    int ch = blockIdx.z;
    int per = Npos / MEAN_CHUNKS;
    const float* hp = h + (size_t)b * Npos * C + (size_t)ch * per * C;
    float s = 0.0f;
    for (int n = 0; n < per; ++n) s += hp[(size_t)n * C + c];
    part[((size_t)ch * nbatch + b) * C + c] = s;
}

__global__ void mean_final(const float* __restrict__ part, float* __restrict__ avg,
                           int Npos, int C) {
    int c = blockIdx.x * 64 + threadIdx.x;
    int b = blockIdx.y;
    int nbatch = gridDim.y;
    float s = 0.0f;
    for (int ch = 0; ch < MEAN_CHUNKS; ++ch)
        s += part[((size_t)ch * nbatch + b) * C + c];
    avg[b * C + c] = s / (float)Npos;
}

__global__ void se_gate(const float* __restrict__ avg,
                        const float* __restrict__ w1, const float* __restrict__ b1,
                        const float* __restrict__ w2, const float* __restrict__ b2,
                        float* __restrict__ g, int C, int inner) {
    int b = blockIdx.x;
    __shared__ float t[128];
    if ((int)threadIdx.x < inner) {
        float s = b1[threadIdx.x];
        for (int c = 0; c < C; ++c) s += avg[b * C + c] * w1[c * inner + threadIdx.x];
        t[threadIdx.x] = silu_f(s);
    }
    __syncthreads();
    for (int j = threadIdx.x; j < C; j += blockDim.x) {
        float s = b2[j];
        for (int i = 0; i < inner; ++i) s += t[i] * w2[i * C + j];
        g[b * C + j] = 1.0f / (1.0f + __expf(-s));
    }
}

__global__ void scale_residual(const float* __restrict__ h, const float* __restrict__ g,
                               const float* __restrict__ res,
                               float* __restrict__ of32, _Float16* __restrict__ of16,
                               int Npos, int C, size_t total) {
    size_t i = (size_t)blockIdx.x * blockDim.x + threadIdx.x;
    if (i >= total) return;
    int c = (int)(i % C);
    int b = (int)((i / C) / Npos);
    float v = h[i] * g[b * C + c] + res[i];
    of32[i] = v;
    of16[i] = (_Float16)v;
}

// ---------------------------------------------------------------------------
// 8) final transpose [b][n][c] -> [b][c][n] into d_out
// ---------------------------------------------------------------------------
__global__ void write_out(const float* __restrict__ x, float* __restrict__ out,
                          int nb, int Npos, int C) {
    size_t total = (size_t)nb * Npos * C;
    size_t i = (size_t)blockIdx.x * blockDim.x + threadIdx.x;
    if (i >= total) return;
    int c = (int)(i % C);
    int n = (int)((i / C) % Npos);
    int b = (int)(i / ((size_t)C * Npos));
    out[((size_t)b * C + c) * Npos + n] = x[i];
}

// ---------------------------------------------------------------------------
// host launcher
// ---------------------------------------------------------------------------
extern "C" void kernel_launch(void* const* d_in, const int* in_sizes, int n_in,
                              void* d_out, int out_size, void* d_ws, size_t ws_size,
                              hipStream_t stream) {
    (void)in_sizes; (void)n_in; (void)out_size; (void)ws_size;
    const int NB = 2, NV = 8192, NF = 16384;
    const int M = NB * NF;               // 32768 faces total
    const int D = 192;                   // codebook dim

    const float* vertices   = (const float*)d_in[0];
    const int*   faces      = (const int*)  d_in[1];
    const float* coor_tab   = (const float*)d_in[2];
    const float* angle_tab  = (const float*)d_in[3];
    const float* area_tab   = (const float*)d_in[4];
    const float* normal_tab = (const float*)d_in[5];
    const float* proj_w     = (const float*)d_in[6];
    const float* proj_b     = (const float*)d_in[7];
    const float* init_w     = (const float*)d_in[8];
    const float* init_b     = (const float*)d_in[9];

    // ---- workspace bump allocator ----
    char* ws = (char*)d_ws;
    size_t off = 0;
    auto alloc = [&](size_t bytes) -> void* {
        void* p = (void*)(ws + off);
        off = (off + bytes + 255) & ~(size_t)255;
        return p;
    };
    int*       idxbuf = (int*)      alloc((size_t)M * 16 * 4);
    _Float16*  feat16 = (_Float16*) alloc((size_t)M * 832 * 2);
    float*     fembed = (float*)    alloc((size_t)M * D * 4);
    float*     vsum   = (float*)    alloc((size_t)NB * NV * D * 4);
    float*     cnt    = (float*)    alloc((size_t)NB * NV * 4);
    _Float16*  gath16 = (_Float16*) alloc((size_t)M * 576 * 2);
    float*     Y      = (float*)    alloc((size_t)M * 384 * 4);
    float*     H32    = (float*)    alloc((size_t)M * 384 * 4);
    _Float16*  H16    = (_Float16*) alloc((size_t)M * 384 * 2);
    float*     Pa32   = (float*)    alloc((size_t)M * 384 * 4);
    _Float16*  Pa16   = (_Float16*) alloc((size_t)M * 384 * 2);
    float*     Pb32   = (float*)    alloc((size_t)M * 384 * 4);
    _Float16*  Pb16   = (_Float16*) alloc((size_t)M * 384 * 2);
    float*     RES32  = (float*)    alloc((size_t)M * 384 * 4);
    float*     avg    = (float*)    alloc((size_t)NB * 384 * 4);
    float*     gate   = (float*)    alloc((size_t)NB * 384 * 4);
    float*     mpart  = (float*)    alloc((size_t)MEAN_CHUNKS * NB * 384 * 4);
    _Float16*  wpool  = (_Float16*) alloc((size_t)8 << 20);
    size_t wcur = 0;
    auto walloc = [&](size_t elems) -> _Float16* {
        _Float16* p = wpool + wcur;
        wcur += (elems + 127) & ~(size_t)127;
        return p;
    };

    auto cdiv = [](size_t a, size_t b) -> unsigned { return (unsigned)((a + b - 1) / b); };

    // ================= encoder =================
    face_feat_idx<<<cdiv(M, 256), 256, 0, stream>>>(vertices, faces, idxbuf, NB, NV, NF);
    build_feat<<<cdiv((size_t)M * 832, 256), 256, 0, stream>>>(
        idxbuf, coor_tab, angle_tab, normal_tab, area_tab, feat16, M);

    // projection GEMM: [M,832] x [832,192]   (Cout 192 = 3 x 64-wide tiles)
    _Float16* wproj = walloc((size_t)832 * 192);
    prep_w_packed<<<cdiv((size_t)832 * 192, 256), 256, 0, stream>>>(proj_w, wproj, 192, 832, 1, 1);
    conv_gemm_wmma<<<dim3(M / 64, D / 64, 1), 128, 0, stream>>>(
        feat16, wproj, proj_b, fembed, M, 832, D, 1, 0);

    // scatter-mean to vertices, gather back per face vertex
    zero_f32<<<cdiv((size_t)NB * NV * D, 256), 256, 0, stream>>>(vsum, (size_t)NB * NV * D);
    zero_f32<<<cdiv((size_t)NB * NV, 256), 256, 0, stream>>>(cnt, (size_t)NB * NV);
    scatter_add<<<cdiv((size_t)M * 3 * D, 256), 256, 0, stream>>>(
        fembed, faces, vsum, cnt, NB, NV, NF, D);
    gather_vfeat<<<cdiv((size_t)M * 3 * D, 256), 256, 0, stream>>>(
        vsum, cnt, faces, gath16, NB, NV, NF, D);

    // ================= decoder =================
    // init conv: 576 -> 128, k=7, pad=3, then SiLU (no pixel norm)
    _Float16* winit = walloc((size_t)7 * 576 * 128);
    prep_w_packed<<<cdiv((size_t)7 * 576 * 128, 256), 256, 0, stream>>>(init_w, winit, 128, 576, 7, 0);
    conv_gemm_wmma<<<dim3(NF / 64, 128 / 64, NB), 128, 0, stream>>>(
        gath16, winit, init_b, Y, NF, 576, 128, 7, 3);
    norm_silu<<<M, 256, 0, stream>>>(Y, Pa32, Pa16, 128, 0);

    float*    cur32 = Pa32;  _Float16* cur16 = Pa16;
    float*    nxt32 = Pb32;  _Float16* nxt16 = Pb16;

    struct Blk { int din, dout, inner; int has_res; int i0; };
    const Blk blks[5] = {
        {128, 128, 32, 0, 10},
        {128, 192, 48, 1, 18},
        {192, 192, 48, 0, 28},
        {192, 256, 64, 1, 36},
        {256, 384, 96, 1, 46},
    };

    for (int bi = 0; bi < 5; ++bi) {
        const Blk& B = blks[bi];
        const float* w1    = (const float*)d_in[B.i0 + 0];
        const float* b1    = (const float*)d_in[B.i0 + 1];
        const float* w2    = (const float*)d_in[B.i0 + 2];
        const float* b2    = (const float*)d_in[B.i0 + 3];
        const float* se_w1 = (const float*)d_in[B.i0 + 4];
        const float* se_b1 = (const float*)d_in[B.i0 + 5];
        const float* se_w2 = (const float*)d_in[B.i0 + 6];
        const float* se_b2 = (const float*)d_in[B.i0 + 7];

        // residual path
        const float* resptr = cur32;
        if (B.has_res) {
            const float* res_w = (const float*)d_in[B.i0 + 8];
            const float* res_b = (const float*)d_in[B.i0 + 9];
            _Float16* wres = walloc((size_t)B.din * B.dout);
            prep_w_packed<<<cdiv((size_t)B.din * B.dout, 256), 256, 0, stream>>>(
                res_w, wres, B.dout, B.din, 1, 0);
            conv_gemm_wmma<<<dim3(NF / 64, B.dout / 64, NB), 128, 0, stream>>>(
                cur16, wres, res_b, RES32, NF, B.din, B.dout, 1, 0);
            resptr = RES32;
        }

        // conv1 + pixelnorm + silu
        _Float16* w1p = walloc((size_t)3 * B.din * B.dout);
        prep_w_packed<<<cdiv((size_t)3 * B.din * B.dout, 256), 256, 0, stream>>>(
            w1, w1p, B.dout, B.din, 3, 0);
        conv_gemm_wmma<<<dim3(NF / 64, B.dout / 64, NB), 128, 0, stream>>>(
            cur16, w1p, b1, Y, NF, B.din, B.dout, 3, 1);
        norm_silu<<<M, 256, 0, stream>>>(Y, H32, H16, B.dout, 1);

        // conv2 + pixelnorm + silu
        _Float16* w2p = walloc((size_t)3 * B.dout * B.dout);
        prep_w_packed<<<cdiv((size_t)3 * B.dout * B.dout, 256), 256, 0, stream>>>(
            w2, w2p, B.dout, B.dout, 3, 0);
        conv_gemm_wmma<<<dim3(NF / 64, B.dout / 64, NB), 128, 0, stream>>>(
            H16, w2p, b2, Y, NF, B.dout, B.dout, 3, 1);
        norm_silu<<<M, 256, 0, stream>>>(Y, H32, H16, B.dout, 1);

        // SE gate (two-stage mean for parallelism)
        mean_partial<<<dim3(B.dout / 64, NB, MEAN_CHUNKS), 64, 0, stream>>>(H32, mpart, NF, B.dout);
        mean_final<<<dim3(B.dout / 64, NB), 64, 0, stream>>>(mpart, avg, NF, B.dout);
        se_gate<<<NB, 128, 0, stream>>>(avg, se_w1, se_b1, se_w2, se_b2, gate, B.dout, B.inner);

        // h * g + res
        size_t total = (size_t)M * B.dout;
        scale_residual<<<cdiv(total, 256), 256, 0, stream>>>(
            H32, gate, resptr, nxt32, nxt16, NF, B.dout, total);

        // swap ping-pong
        float* t32 = cur32; cur32 = nxt32; nxt32 = t32;
        _Float16* t16 = cur16; cur16 = nxt16; nxt16 = t16;
    }

    // [b][n][384] -> [b][384][n]
    write_out<<<cdiv((size_t)M * 384, 256), 256, 0, stream>>>(
        cur32, (float*)d_out, NB, NF, 384);
}
